// RelationAwareAttention_49555332661577
// MI455X (gfx1250) — matrix operand order repaired and checked by reading, not compiled, not run on hardware
//
#include <hip/hip_runtime.h>
#include <math.h>

typedef __attribute__((ext_vector_type(16))) _Float16 v16h;
typedef __attribute__((ext_vector_type(8)))  _Float16 v8h;
typedef __attribute__((ext_vector_type(8)))  float    v8f;
typedef __attribute__((ext_vector_type(4)))  float    v4f;

#define B_  2
#define H_  8
#define L_  2048
#define DK_ 64
#define WAVES 8
#define ROWS_PER_WAVE 16
#define ROWS_PER_WG (WAVES * ROWS_PER_WAVE)   // 128 rows per workgroup
#define SCALE 0.07216878364870322f            // 1/sqrt(3*64)

__global__ __launch_bounds__(256)
void raa_fused_kernel(const float* __restrict__ Q, const float* __restrict__ K,
                      const float* __restrict__ V, const unsigned char* __restrict__ Mask,
                      const float* __restrict__ Tree, const float* __restrict__ Leaf,
                      float* __restrict__ OutO, float* __restrict__ OutP)
{
    const int lane = threadIdx.x & 31;
    const int wave = threadIdx.x >> 5;
    const int n    = lane & 15;   // column within 16-wide tile (B/C/D layout)
    const int half = lane >> 4;   // half-wave group
    const int m    = n;           // row within tile for A layout

    const int bh = blockIdx.y;        // 0..15
    const int b  = bh / H_;
    const int i0 = blockIdx.x * ROWS_PER_WG + wave * ROWS_PER_WAVE;

    const size_t qk_base   = (size_t)bh * L_ * DK_;
    const size_t bias_base = (size_t)bh * L_ * L_;
    const size_t mask_base = (size_t)b  * L_ * L_;

    // Per-wave P tile: 16 rows x (32 used + 8 pad) f16; 80B row stride keeps b128 loads aligned.
    __shared__ __align__(16) _Float16 tileP[WAVES][16][40];

    // ---- Load Q rows once into A-matrix layout (two K=32 slices) ----
    // A operand in this code's k order (16x32): lane m = lane%16; element e of slice s maps to
// Kdim = s*32 + 16*half + e  -> one contiguous 16-float run.
    v16h aQ[2];
    {
        const float* qrow = Q + qk_base + (size_t)(i0 + m) * DK_;
        #pragma unroll
        for (int s = 0; s < 2; ++s) {
            #pragma unroll
            for (int g = 0; g < 2; ++g) {
                const float* p = qrow + s * 32 + half * 16 + g * 8;
                v4f f0 = *(const v4f*)(p);
                v4f f1 = *(const v4f*)(p + 4);
                #pragma unroll
                for (int e = 0; e < 4; ++e) {
                    aQ[s][g * 8 + e]     = (_Float16)f0[e];
                    aQ[s][g * 8 + 4 + e] = (_Float16)f1[e];
                }
            }
        }
    }

    float rm[8], rs[8];
    #pragma unroll
    for (int r = 0; r < 8; ++r) { rm[r] = -3.0e38f; rs[r] = 0.0f; }

    // ---- Pass 1: S = (QK^T + tree + leaf)*scale, masked; write S; online max/sum ----
    for (int jt = 0; jt < L_ / 16; ++jt) {
        const int j0 = jt * 16;
        // B operand, same k order as A (32x16): lane n = lane%16; element e -> Kdim = s*32 + 16*half + e
        // -> one contiguous 16-float run of K[j0+n][...]
        v16h bK[2];
        {
            const float* krow = K + qk_base + (size_t)(j0 + n) * DK_;
            #pragma unroll
            for (int s = 0; s < 2; ++s) {
                const float* p = krow + s * 32 + half * 16;
                #pragma unroll
                for (int q4 = 0; q4 < 4; ++q4) {
                    v4f f = *(const v4f*)(p + q4 * 4);
                    #pragma unroll
                    for (int e = 0; e < 4; ++e) bK[s][q4 * 4 + e] = (_Float16)f[e];
                }
            }
        }

        // Batch ALL streaming loads unconditionally (non-temporal: read-once data),
        // so the mask select below lowers to v_cndmask, not divergent branches.
        float tv[8], lv[8];
        unsigned char mk[8];
        #pragma unroll
        for (int r = 0; r < 8; ++r) {
            const int row = i0 + r + 8 * half;
            const int col = j0 + n;
            const size_t off = bias_base + (size_t)row * L_ + col;
            tv[r] = __builtin_nontemporal_load(&Tree[off]);
            lv[r] = __builtin_nontemporal_load(&Leaf[off]);
            mk[r] = __builtin_nontemporal_load(&Mask[mask_base + (size_t)row * L_ + col]);
        }

        v8f acc = {};
        acc = __builtin_amdgcn_wmma_f32_16x16x32_f16(false, aQ[0], false, bK[0],
                                                     (short)0, acc, false, false);
        acc = __builtin_amdgcn_wmma_f32_16x16x32_f16(false, aQ[1], false, bK[1],
                                                     (short)0, acc, false, false);

        // C/D layout: lane holds column j0+n for rows (r + 8*half), r = 0..7
        #pragma unroll
        for (int r = 0; r < 8; ++r) {
            const int row = i0 + r + 8 * half;
            const int col = j0 + n;
            const size_t off = bias_base + (size_t)row * L_ + col;
            float svv = (acc[r] + tv[r] + lv[r]) * SCALE;
            float sv  = (mk[r] != 0) ? -1.0e9f : svv;     // pure value select
            OutP[off] = sv;                  // raw score; re-read from L2 in pass 2
            float mnew = fmaxf(rm[r], sv);
            rs[r] = rs[r] * __expf(rm[r] - mnew) + __expf(sv - mnew);
            rm[r] = mnew;
        }
    }

    // ---- Cross-lane softmax reduction within each 16-lane half-group ----
    #pragma unroll
    for (int r = 0; r < 8; ++r) {
        #pragma unroll
        for (int off = 1; off < 16; off <<= 1) {
            float om = __shfl_xor(rm[r], off, 32);
            float os = __shfl_xor(rs[r], off, 32);
            float mnew = fmaxf(rm[r], om);
            rs[r] = rs[r] * __expf(rm[r] - mnew) + os * __expf(om - mnew);
            rm[r] = mnew;
        }
        rs[r] = 1.0f / rs[r];   // rs now holds 1/denominator
    }

    // Make our score stores visible before each lane's re-reads of its own scores in pass 2.
    asm volatile("s_wait_storecnt 0" ::: "memory");

    // ---- Pass 2: P = exp(S - max)/denom (write back), O = P * V via WMMA ----
    v8f accO[4] = {};   // output N-tiles: cols 0-15, 16-31, 32-47, 48-63
    for (int jt = 0; jt < L_ / 32; ++jt) {
        const int j0 = jt * 32;
        // Normalize two 16-col subtiles (coalesced), stash f16 tile to LDS.
        #pragma unroll
        for (int st = 0; st < 2; ++st) {
            const int col = j0 + st * 16 + n;
            float sv[8];
            #pragma unroll
            for (int r = 0; r < 8; ++r) {
                const int row = i0 + r + 8 * half;
                sv[r] = OutP[bias_base + (size_t)row * L_ + col];  // L2 hit
            }
            #pragma unroll
            for (int r = 0; r < 8; ++r) {
                const int row = i0 + r + 8 * half;
                const size_t off = bias_base + (size_t)row * L_ + col;
                float p = __expf(sv[r] - rm[r]) * rs[r];
                __builtin_nontemporal_store(p, &OutP[off]);        // final P, never re-read
                tileP[wave][r + 8 * half][st * 16 + n] = (_Float16)p;
            }
        }
        __syncthreads();

        // A-matrix layout from LDS: two aligned b128 reads per lane.
        v16h aP;
        {
            const _Float16* rowp = &tileP[wave][m][0];
            v8h lo = *(const v8h*)(rowp + half * 16);       // K = 16*half + 0..7
            v8h hi = *(const v8h*)(rowp + half * 16 + 8);   // K = 16*half + 8..15
            #pragma unroll
            for (int e = 0; e < 8; ++e) { aP[e] = lo[e]; aP[8 + e] = hi[e]; }
        }

        // V in B layout, 4 N-tiles; reads are lane-coalesced per element row.
        #pragma unroll
        for (int nt = 0; nt < 4; ++nt) {
            v16h bV;
            const float* vcol = V + qk_base + (size_t)(j0 + 16 * half) * DK_ + nt * 16 + n;
            #pragma unroll
            for (int e = 0; e < 16; ++e) bV[e] = (_Float16)vcol[(size_t)e * DK_];
            accO[nt] = __builtin_amdgcn_wmma_f32_16x16x32_f16(false, aP, false, bV,
                                                              (short)0, accO[nt], false, false);
        }
        __syncthreads();
    }

    // ---- Write attn_sum (C/D layout -> row-major) ----
    #pragma unroll
    for (int nt = 0; nt < 4; ++nt) {
        #pragma unroll
        for (int r = 0; r < 8; ++r) {
            const int row = i0 + r + 8 * half;
            OutO[qk_base + (size_t)row * DK_ + nt * 16 + n] = accO[nt][r];
        }
    }
}

extern "C" void kernel_launch(void* const* d_in, const int* in_sizes, int n_in,
                              void* d_out, int out_size, void* d_ws, size_t ws_size,
                              hipStream_t stream) {
    (void)in_sizes; (void)n_in; (void)d_ws; (void)ws_size; (void)out_size;
    const float*         Q    = (const float*)d_in[0];
    const float*         K    = (const float*)d_in[1];
    const float*         V    = (const float*)d_in[2];
    const unsigned char* Mask = (const unsigned char*)d_in[3];   // jnp.bool_ -> 1 byte
    const float*         Tree = (const float*)d_in[4];
    const float*         Leaf = (const float*)d_in[5];

    float* OutO = (float*)d_out;                               // (B,H,L,DK)
    float* OutP = OutO + (size_t)B_ * H_ * L_ * DK_;           // (B,H,L,L)

    dim3 grid(L_ / ROWS_PER_WG, B_ * H_);   // (16, 16)
    dim3 block(256);                        // 8 waves (wave32)
    raa_fused_kernel<<<grid, block, 0, stream>>>(Q, K, V, Mask, Tree, Leaf, OutO, OutP);
}
